// GotenNetCore_52063593562429
// MI455X (gfx1250) — matrix operand order repaired and checked by reading, not compiled, not run on hardware
//
#include <hip/hip_runtime.h>
#include <math.h>

#define NN 10000
#define EE 160000
#define HH 256
#define LL 4

typedef float  v8f   __attribute__((ext_vector_type(8)));
typedef __bf16 v16bf __attribute__((ext_vector_type(16)));
typedef __bf16 v8bf  __attribute__((ext_vector_type(8)));

static __device__ __forceinline__ v8f wmma_bf16(v16bf a, v16bf b, v8f c) {
  return __builtin_amdgcn_wmma_f32_16x16x32_bf16(false, a, false, b, (short)0, c, false, false);
}

// A-fragment (16x32 bf16) per-lane load from a row-major row: element j maps to
// k = base + (j&7) + 16*(j>>3); i.e. two contiguous 8-half chunks at p and p+16.
static __device__ __forceinline__ v16bf ld_afrag(const __bf16* p) {
  union { v16bf v; v8bf h[2]; } u;
  u.h[0] = *(const v8bf*)p;
  u.h[1] = *(const v8bf*)(p + 16);
  return u.v;
}

static __device__ __forceinline__ v8f splat8(float x) {
  v8f c = {x, x, x, x, x, x, x, x};
  return c;
}

// silu with hardware reciprocal (v_rcp_f32): avoids the IEEE div expansion.
static __device__ __forceinline__ float siluf(float x) {
  return x * __builtin_amdgcn_rcpf(1.0f + __expf(-x));
}

static __device__ __forceinline__ void atomAdd(float* p, float v) {
  __hip_atomic_fetch_add(p, v, __ATOMIC_RELAXED, __HIP_MEMORY_SCOPE_AGENT);
}

static __device__ __forceinline__ void atomAddLds(float* p, float v) {
  __hip_atomic_fetch_add(p, v, __ATOMIC_RELAXED, __HIP_MEMORY_SCOPE_WORKGROUP);
}

// ---------------------------------------------------------------------------
// Pack a row-major [Ksrc, Nc] f32 weight into bf16 B-fragments:
// frag index t = (kt*NT + nt)*32 + lane ; half j -> k = kt*32 + (j&7) + 16*(j>>3) + 8*(lane>=16),
// n = nt*16 + (lane&15).  Zero-pads k >= Ksrc.
__global__ void pack_b_kernel(const float* __restrict__ src, __bf16* __restrict__ dst,
                              int Ksrc, int NT, int Nc, int total) {
  int t = blockIdx.x * blockDim.x + threadIdx.x;
  if (t >= total) return;
  int lane = t & 31;
  int nt = (t >> 5) % NT;
  int kt = (t >> 5) / NT;
  int lanehi = lane >> 4, lanelo = lane & 15;
  v16bf o;
#pragma unroll
  for (int j = 0; j < 16; ++j) {
    int k = kt * 32 + (j & 7) + 16 * (j >> 3) + 8 * lanehi;
    int n = nt * 16 + lanelo;
    float v = (k < Ksrc) ? src[(size_t)k * Nc + n] : 0.0f;
    o[j] = (__bf16)v;
  }
  *(v16bf*)(dst + (size_t)t * 16) = o;
}

// edge_attr = exp(coeff*(d-offset_k)^2) * cosine_cutoff(d), packed as WMMA A
// fragments: per 16-edge tile, 2 K-fragments (K=64 padded from 50).
__global__ void pack_eattr_kernel(const float* __restrict__ ediff, __bf16* __restrict__ dst) {
  int t = blockIdx.x * blockDim.x + threadIdx.x;  // ((tile*2+frag)*32 + lane)
  if (t >= (EE / 16) * 2 * 32) return;
  int lane = t & 31, frag = (t >> 5) & 1, tile = t >> 6;
  int lanehi = lane >> 4, lanelo = lane & 15;
  int e = tile * 16 + lanelo;  // A row m = lane&15
  float dd = ediff[e];
  float cut = (dd < 5.0f) ? 0.5f * (__cosf(dd * 0.628318530717958f) + 1.0f) : 0.0f;
  const float step = 5.0f / 49.0f;
  const float coeff = -0.5f / (step * step);
  v16bf o;
#pragma unroll
  for (int j = 0; j < 16; ++j) {
    int k = frag * 32 + (j & 7) + 16 * (j >> 3) + 8 * lanehi;
    float val = 0.0f;
    if (k < 50) {
      float x = dd - (float)k * step;
      val = __expf(coeff * x * x) * cut;
    }
    o[j] = (__bf16)val;
  }
  *(v16bf*)(dst + (size_t)t * 16) = o;
}

__global__ void init_kernel(const int* __restrict__ z, const float* __restrict__ emb,
                            float* __restrict__ s, float* __restrict__ v) {
  size_t i = (size_t)blockIdx.x * blockDim.x + threadIdx.x;
  const size_t nh = (size_t)NN * HH;
  if (i < nh) s[i] = emb[(size_t)z[i >> 8] * HH + (i & 255)];
  if (i < nh * 3) v[i] = 0.0f;
}

// ---------------------------------------------------------------------------
// Stage A: edge MLP (2 GEMMs via WMMA) fused with message formation + scatter.
// One wave per 16-edge tile; 4 waves per block.  B fragments are staged in LDS
// once per block (4x reduction in L2 weight traffic).
__global__ __launch_bounds__(128) void edge_kernel(
    const float* __restrict__ sSrc, const float* __restrict__ vSrc,
    float* __restrict__ sDst, float* __restrict__ vDst,
    const __bf16* __restrict__ eattr,
    const __bf16* __restrict__ w1f, const float* __restrict__ b1,
    const __bf16* __restrict__ w2f, const float* __restrict__ b2,
    const int* __restrict__ ei,
    const float* __restrict__ evec, const float* __restrict__ ediff) {
  __shared__ __bf16 h1[4][16 * 256];  // 32KB
  __shared__ v16bf bstage[256];       // 8KB: staged B fragments (max 8 per nt)
  const int tid = threadIdx.x;
  const int lane = tid & 31;
  const int wave = tid >> 5;
  const int tile = blockIdx.x * 4 + wave;  // exact: 10000 tiles
  const int lanehi = lane >> 4, lanelo = lane & 15;
  __bf16* hrow = &h1[wave][0];

  v16bf a0 = *(const v16bf*)(eattr + ((size_t)tile * 2 + 0) * 512 + lane * 16);
  v16bf a1 = *(const v16bf*)(eattr + ((size_t)tile * 2 + 1) * 512 + lane * 16);

  // GEMM1: [16,64] x [64,256] -> silu -> h1 (bf16 row-major in LDS)
  for (int nt = 0; nt < 16; ++nt) {
    __syncthreads();
    if (tid < 64)  // 2 fragments = 64 v16bf
      bstage[tid] = *(const v16bf*)(w1f + (size_t)(((tid >> 5) * 16 + nt) * 32 + (tid & 31)) * 16);
    __syncthreads();
    v8f c = splat8(b1[nt * 16 + lanelo]);
    c = wmma_bf16(a0, bstage[lane], c);
    c = wmma_bf16(a1, bstage[32 + lane], c);
#pragma unroll
    for (int r = 0; r < 8; ++r) {
      int m = r + 8 * lanehi;
      hrow[m * 256 + nt * 16 + lanelo] = (__bf16)siluf(c[r]);
    }
  }

  // Preload all 8 A-fragments of h1 for GEMM2 (K=256).
  v16bf a2[8];
#pragma unroll
  for (int kk = 0; kk < 8; ++kk)
    a2[kk] = ld_afrag(&hrow[lanelo * 256 + kk * 32 + 8 * lanehi]);

  // Per-lane edge metadata for the 8 C-rows this lane owns.
  int srcs[8], dsts[8];
  float evx[8], evy[8], evz[8];
#pragma unroll
  for (int r = 0; r < 8; ++r) {
    int e = tile * 16 + r + 8 * lanehi;
    srcs[r] = ei[e] * HH;
    dsts[r] = ei[EE + e] * HH;
    float inv = __builtin_amdgcn_rcpf(ediff[e] + 1e-6f);
    evx[r] = evec[e * 3 + 0] * inv;
    evy[r] = evec[e * 3 + 1] * inv;
    evz[r] = evec[e * 3 + 2] * inv;
  }

  // GEMM2: [16,256] x [256,768]; consume W tiles directly (thirds commute under +).
  for (int nt = 0; nt < 48; ++nt) {
    __syncthreads();
#pragma unroll
    for (int i = 0; i < 2; ++i) {  // 8 fragments = 256 v16bf, 128 threads
      int c8 = tid + i * 128;
      bstage[c8] = *(const v16bf*)(w2f + (size_t)(((c8 >> 5) * 48 + nt) * 32 + (c8 & 31)) * 16);
    }
    __syncthreads();
    v8f c = splat8(b2[nt * 16 + lanelo]);
#pragma unroll
    for (int kk = 0; kk < 8; ++kk)
      c = wmma_bf16(a2[kk], bstage[kk * 32 + lane], c);
    const int third = nt >> 4;  // 0:W_s  1:W_v1  2:W_v2
    const int h = (nt & 15) * 16 + lanelo;
#pragma unroll
    for (int r = 0; r < 8; ++r) {
      float w = c[r];
      int si = srcs[r] + h;
      int di = dsts[r] + h;
      if (third == 0) {
        atomAdd(&sDst[di], sSrc[si] * w);
      } else if (third == 1) {
        atomAdd(&vDst[di * 3 + 0], vSrc[si * 3 + 0] * w);
        atomAdd(&vDst[di * 3 + 1], vSrc[si * 3 + 1] * w);
        atomAdd(&vDst[di * 3 + 2], vSrc[si * 3 + 2] * w);
      } else {
        float sj = sSrc[si] * w;
        atomAdd(&vDst[di * 3 + 0], sj * evx[r]);
        atomAdd(&vDst[di * 3 + 1], sj * evy[r]);
        atomAdd(&vDst[di * 3 + 2], sj * evz[r]);
      }
    }
  }
}

// ---------------------------------------------------------------------------
// Stage B: GeometricUpdate.  Cooperative 8-wave workgroup per 16-node tile:
//  - v_proj: 6 waves split (d, j-half); dot/norm2 combined via ds_add_f32;
//    remaining 2 waves idle; all 8 waves prefill the s-half of s_cat beforehand.
//  - up1/up2/sp N-tiles are work-split across all 8 waves with barriers.
__global__ __launch_bounds__(256) void geom_kernel(
    float* __restrict__ s, float* __restrict__ v,
    const __bf16* __restrict__ vwf,
    const __bf16* __restrict__ u1f, const float* __restrict__ ub1,
    const __bf16* __restrict__ u2f, const float* __restrict__ ub2,
    const __bf16* __restrict__ spf, const float* __restrict__ spb,
    float* __restrict__ vu_ws) {
  __shared__ float dotb[4096];   // 16KB: dot(v_u, v_v)                (live whole kernel)
  __shared__ float regB[4096];   // 16KB: norm2 f32 -> hbuf bf16 [0..8K) + asv bf16 [8K..16K)
  __shared__ __bf16 regC[8192];  // 16KB: s_cat bf16 [16][512] -> ds2 bf16 [16][256]
  const int tid = threadIdx.x;
  const int lane = tid & 31;
  const int wave = tid >> 5;
  const int tile = blockIdx.x;  // 625 exact
  const int lanehi = lane >> 4, lanelo = lane & 15;
  __bf16* scat = regC;

  // P0: zero accumulators; prefill s-half of s_cat (independent of v_proj).
  for (int idx = tid; idx < 4096; idx += 256) {
    dotb[idx] = 0.0f;
    regB[idx] = 0.0f;
    int m = idx >> 8, h = idx & 255;
    scat[m * 512 + h] = (__bf16)s[(size_t)(tile * 16 + m) * 256 + h];
  }
  __syncthreads();

  // P1: v_proj = v[:, :, d] @ vec_w.  wave w in [0,6): d = w>>1, j-half = w&1.
  // v_u tile j pairs with v_v tile j+16 for dot/norm2 (ds_add_f32 across d-waves).
  if (wave < 6) {
    const int d = wave >> 1;
    const int jbase = (wave & 1) * 8;
    const int nodeA = tile * 16 + lanelo;
    v16bf a[8];
#pragma unroll
    for (int kk = 0; kk < 8; ++kk) {
      v16bf t;
#pragma unroll
      for (int j = 0; j < 16; ++j) {
        int k = kk * 32 + (j & 7) + 16 * (j >> 3) + 8 * lanehi;
        t[j] = (__bf16)v[((size_t)nodeA * 256 + k) * 3 + d];
      }
      a[kk] = t;
    }
    for (int jj = 0; jj < 8; ++jj) {
      int j = jbase + jj;
      v8f cu = splat8(0.0f), cv = splat8(0.0f);
#pragma unroll
      for (int kk = 0; kk < 8; ++kk)
        cu = wmma_bf16(a[kk], *(const v16bf*)(vwf + (size_t)((kk * 32 + j) * 32 + lane) * 16), cu);
#pragma unroll
      for (int kk = 0; kk < 8; ++kk)
        cv = wmma_bf16(a[kk], *(const v16bf*)(vwf + (size_t)((kk * 32 + j + 16) * 32 + lane) * 16), cv);
#pragma unroll
      for (int r = 0; r < 8; ++r) {
        int m = r + 8 * lanehi;
        int h = j * 16 + lanelo;
        int idx = m * 256 + h;
        float pu = cu[r], pv = cv[r];
        atomAddLds(&dotb[idx], pu * pv);
        atomAddLds(&regB[idx], pv * pv);
        vu_ws[(size_t)d * ((size_t)NN * 256) + (size_t)(tile * 16 + m) * 256 + h] = pu;
      }
    }
  }
  __syncthreads();

  // P3: vnorm half of s_cat.
  for (int idx = tid; idx < 4096; idx += 256) {
    int m = idx >> 8, h = idx & 255;
    scat[m * 512 + 256 + h] = (__bf16)sqrtf(regB[idx]);
  }
  __syncthreads();

  // P4: up1: [16,512] x [512,256] -> silu -> hbuf (regB reuse; norm2 dead).
  __bf16* hbuf = (__bf16*)regB;
#pragma unroll
  for (int t = 0; t < 2; ++t) {
    int nt = wave * 2 + t;
    v8f c = splat8(ub1[nt * 16 + lanelo]);
#pragma unroll
    for (int kk = 0; kk < 16; ++kk) {
      v16bf aa = ld_afrag(&scat[lanelo * 512 + kk * 32 + 8 * lanehi]);
      v16bf bb = *(const v16bf*)(u1f + (size_t)((kk * 16 + nt) * 32 + lane) * 16);
      c = wmma_bf16(aa, bb, c);
    }
#pragma unroll
    for (int r = 0; r < 8; ++r) {
      int m = r + 8 * lanehi;
      hbuf[m * 256 + nt * 16 + lanelo] = (__bf16)siluf(c[r]);
    }
  }
  __syncthreads();

  // P5a: up2 tiles nt=0..31: a_vv -> dv2 (global), a_sv -> LDS stash.
  __bf16* asv = ((__bf16*)regB) + 4096;  // bytes 8K..16K of regB
  __bf16* ds2 = regC;                    // reuses scat after up1
#pragma unroll
  for (int t = 0; t < 4; ++t) {
    int nt = wave * 4 + t;
    v8f c = splat8(ub2[nt * 16 + lanelo]);
#pragma unroll
    for (int kk = 0; kk < 8; ++kk) {
      v16bf aa = ld_afrag(&hbuf[lanelo * 256 + kk * 32 + 8 * lanehi]);
      v16bf bb = *(const v16bf*)(u2f + (size_t)((kk * 48 + nt) * 32 + lane) * 16);
      c = wmma_bf16(aa, bb, c);
    }
    const int third = nt >> 4;
    const int h = (nt & 15) * 16 + lanelo;
#pragma unroll
    for (int r = 0; r < 8; ++r) {
      int m = r + 8 * lanehi;
      int node = tile * 16 + m;
      int idx = m * 256 + h;
      float x = c[r];
      if (third == 0) {
#pragma unroll
        for (int d = 0; d < 3; ++d) {
          size_t vi = ((size_t)node * 256 + h) * 3 + d;
          v[vi] += vu_ws[(size_t)d * ((size_t)NN * 256) + (size_t)node * 256 + h] * x;
        }
      } else {
        asv[idx] = (__bf16)x;
      }
    }
  }
  __syncthreads();

  // P5b: up2 tiles nt=32..47: ds2 = a_ss + dot*a_sv  (into regC; scat dead).
#pragma unroll
  for (int t = 0; t < 2; ++t) {
    int nt = 32 + wave * 2 + t;
    v8f c = splat8(ub2[nt * 16 + lanelo]);
#pragma unroll
    for (int kk = 0; kk < 8; ++kk) {
      v16bf aa = ld_afrag(&hbuf[lanelo * 256 + kk * 32 + 8 * lanehi]);
      v16bf bb = *(const v16bf*)(u2f + (size_t)((kk * 48 + nt) * 32 + lane) * 16);
      c = wmma_bf16(aa, bb, c);
    }
    const int h = (nt & 15) * 16 + lanelo;
#pragma unroll
    for (int r = 0; r < 8; ++r) {
      int m = r + 8 * lanehi;
      int idx = m * 256 + h;
      ds2[idx] = (__bf16)(c[r] + dotb[idx] * (float)asv[idx]);
    }
  }
  __syncthreads();

  // P6: sp: s += ds2 @ sp_w + sp_b
#pragma unroll
  for (int t = 0; t < 2; ++t) {
    int nt = wave * 2 + t;
    v8f c = splat8(spb[nt * 16 + lanelo]);
#pragma unroll
    for (int kk = 0; kk < 8; ++kk) {
      v16bf aa = ld_afrag(&ds2[lanelo * 256 + kk * 32 + 8 * lanehi]);
      v16bf bb = *(const v16bf*)(spf + (size_t)((kk * 16 + nt) * 32 + lane) * 16);
      c = wmma_bf16(aa, bb, c);
    }
#pragma unroll
    for (int r = 0; r < 8; ++r) {
      int m = r + 8 * lanehi;
      s[(size_t)(tile * 16 + m) * 256 + nt * 16 + lanelo] += c[r];
    }
  }
}

// ---------------------------------------------------------------------------
extern "C" void kernel_launch(void* const* d_in, const int* in_sizes, int n_in,
                              void* d_out, int out_size, void* d_ws, size_t ws_size,
                              hipStream_t stream) {
  (void)in_sizes; (void)n_in; (void)out_size;
  const int*   z     = (const int*)d_in[0];
  const int*   ei    = (const int*)d_in[2];
  const float* evec  = (const float*)d_in[3];
  const float* ediff = (const float*)d_in[4];
  const float* emb   = (const float*)d_in[5];
  const float* iw1   = (const float*)d_in[6];
  const float* ib1   = (const float*)d_in[7];
  const float* iw2   = (const float*)d_in[8];
  const float* ib2   = (const float*)d_in[9];
  const float* vw    = (const float*)d_in[10];
  const float* uw1   = (const float*)d_in[11];
  const float* ub1   = (const float*)d_in[12];
  const float* uw2   = (const float*)d_in[13];
  const float* ub2   = (const float*)d_in[14];
  const float* spw   = (const float*)d_in[15];
  const float* spb   = (const float*)d_in[16];

  const size_t NH = (size_t)NN * HH;            // 2,560,000
  const size_t SV_BYTES = NH * 4ull * 4ull;     // s + v of one buffer set
  char* wsb = (char*)d_ws;

  const size_t OFF_EATTR   = SV_BYTES;
  const size_t EATTR_BYTES = (size_t)(EE / 16) * 2 * 512 * 2;
  const size_t OFF_WF      = OFF_EATTR + EATTR_BYTES;
  const size_t LAYER_WF    = 32768 + 393216 + 262144 + 262144 + 393216 + 131072;
  const size_t OFF_VU      = OFF_WF + 4 * LAYER_WF;
  const size_t NEED        = OFF_VU + NH * 3 * 4;
  if (ws_size < NEED) return;

  float* wsS  = (float*)wsb;
  float* wsV  = wsS + NH;
  float* outS = (float*)d_out;
  float* outV = outS + NH;
  __bf16* eattr = (__bf16*)(wsb + OFF_EATTR);
  float*  vuws  = (float*)(wsb + OFF_VU);

  {  // edge feature fragments (layer-invariant)
    int total = (EE / 16) * 2 * 32;
    pack_eattr_kernel<<<(total + 255) / 256, 256, 0, stream>>>(ediff, eattr);
  }

  for (int l = 0; l < LL; ++l) {  // weight fragments
    char* base = wsb + OFF_WF + (size_t)l * LAYER_WF;
    __bf16* w1f = (__bf16*)(base + 0);
    __bf16* w2f = (__bf16*)(base + 32768);
    __bf16* vwf = (__bf16*)(base + 425984);
    __bf16* u1f = (__bf16*)(base + 688128);
    __bf16* u2f = (__bf16*)(base + 950272);
    __bf16* spf = (__bf16*)(base + 1343488);
    auto pack = [&](const float* src, __bf16* dst, int Ksrc, int KT, int NT, int Nc) {
      int total = KT * NT * 32;
      pack_b_kernel<<<(total + 255) / 256, 256, 0, stream>>>(src, dst, Ksrc, NT, Nc, total);
    };
    pack(iw1 + (size_t)l * 50 * 256,  w1f, 50,   2, 16, 256);
    pack(iw2 + (size_t)l * 256 * 768, w2f, 256,  8, 48, 768);
    pack(vw  + (size_t)l * 256 * 512, vwf, 256,  8, 32, 512);
    pack(uw1 + (size_t)l * 512 * 256, u1f, 512, 16, 16, 256);
    pack(uw2 + (size_t)l * 256 * 768, u2f, 256,  8, 48, 768);
    pack(spw + (size_t)l * 256 * 256, spf, 256,  8, 16, 256);
  }

  {  // s = emb[z], v = 0  into d_out (set1); parity puts final layer back in d_out
    size_t total = NH * 3;
    init_kernel<<<(int)((total + 255) / 256), 256, 0, stream>>>(z, emb, outS, outV);
  }

  for (int l = 0; l < LL; ++l) {
    float* srcS = (l & 1) ? wsS : outS;
    float* dstS = (l & 1) ? outS : wsS;
    float* srcV = (l & 1) ? wsV : outV;
    float* dstV = (l & 1) ? outV : wsV;
    hipMemcpyAsync(dstS, srcS, SV_BYTES, hipMemcpyDeviceToDevice, stream);

    char* base = wsb + OFF_WF + (size_t)l * LAYER_WF;
    __bf16* w1f = (__bf16*)(base + 0);
    __bf16* w2f = (__bf16*)(base + 32768);
    __bf16* vwf = (__bf16*)(base + 425984);
    __bf16* u1f = (__bf16*)(base + 688128);
    __bf16* u2f = (__bf16*)(base + 950272);
    __bf16* spf = (__bf16*)(base + 1343488);

    edge_kernel<<<EE / 16 / 4, 128, 0, stream>>>(
        srcS, srcV, dstS, dstV, eattr,
        w1f, ib1 + (size_t)l * 256, w2f, ib2 + (size_t)l * 768,
        ei, evec, ediff);

    geom_kernel<<<NN / 16, 256, 0, stream>>>(
        dstS, dstV, vwf,
        u1f, ub1 + (size_t)l * 256,
        u2f, ub2 + (size_t)l * 768,
        spf, spb + (size_t)l * 256, vuws);
  }
}
